// SAGPooling_44178033606816
// MI455X (gfx1250) — compile-verified, matrix operand order checked by reading, not run on hardware
//
#include <hip/hip_runtime.h>
#include <stdint.h>

// CDNA5 / gfx1250 SAGPooling pipeline.
// Bandwidth-bound (~1.7 GB total HBM traffic => ~75us floor @ 23.3 TB/s).
// The x@W score GEMV runs on V_WMMA_F32_16X16X4_F32 (fp32-exact, free under
// the load-bound roofline); everything else is streaming / atomic / radix-sort.

typedef __attribute__((ext_vector_type(2)))  float    v2f;
typedef __attribute__((ext_vector_type(8)))  float    v8f;
typedef __attribute__((ext_vector_type(16))) _Float16 v16h;

#define RS_BLOCK 256
#define RS_ITEMS 8
#define RS_TILE  (RS_BLOCK * RS_ITEMS)

// ---------------------------------------------------------------- init ----
__global__ void init_kernel(unsigned* __restrict__ cnt,
                            unsigned long long* __restrict__ aggI,
                            int* __restrict__ new_idx, int N) {
  int i = blockIdx.x * blockDim.x + threadIdx.x;
  if (i < N) { cnt[i] = 0u; aggI[i] = 0ull; new_idx[i] = -1; }
}

// ----------------------------------------------------- xw = x @ W (WMMA) ----
// One wave computes scores for 16 rows. A-frag: 16x4 fp32 tile of x
// (lane m=lane&15 holds row m; half-wave selects K pair 0/1 vs 2/3).
// B-frag: W chunk in column 0 only. W is loaded UNCONDITIONALLY (uniform,
// cache-hot) and masked with v_cndmask — no exec-mask branching in the hot
// loop (round-0 codegen produced s_and_saveexec per W load).
// D column 0 lands in c[0..7] of lane 0 (M=0..7) and lane 16 (M=8..15).
__global__ void xw_score_kernel(const float* __restrict__ x,
                                const float* __restrict__ W,
                                float* __restrict__ xw, int N, int C) {
  const int wave = blockIdx.x * (blockDim.x >> 5) + (threadIdx.x >> 5);
  const int row0 = wave * 16;
  if (row0 >= N) return;
  const int lane = threadIdx.x & 31;
  const int m    = lane & 15;
#if defined(__gfx1250__) && __has_builtin(__builtin_amdgcn_wmma_f32_16x16x4_f32)
  const int  kb   = (lane >> 4) << 1;   // K base within 4-wide step: 0 or 2
  const bool col0 = (m == 0);
  const float* xr = x + (size_t)(row0 + m) * (size_t)C;
  v8f c = {};
#pragma unroll 4
  for (int k0 = 0; k0 < C; k0 += 4) {
    v2f a  = *(const v2f*)(xr + k0 + kb);   // A[m, k0+kb], A[m, k0+kb+1]
    v2f wv = *(const v2f*)(W  + k0 + kb);   // unconditional, branchless
    v2f bw;
    bw.x = col0 ? wv.x : 0.0f;              // v_cndmask, no exec divergence
    bw.y = col0 ? wv.y : 0.0f;
    c = __builtin_amdgcn_wmma_f32_16x16x4_f32(false, a, false, bw,
                                              (short)0, c, false, false);
  }
  if (lane == 0) {
#pragma unroll
    for (int r = 0; r < 8; ++r) xw[row0 + r] = c[r];
  } else if (lane == 16) {
#pragma unroll
    for (int r = 0; r < 8; ++r) xw[row0 + 8 + r] = c[r];
  }
#elif defined(__gfx1250__) && __has_builtin(__builtin_amdgcn_wmma_f32_16x16x32_f16)
  // Fallback: codegen-confirmed f16 WMMA; ISA 7.12.2 16-bit A 16x32 layout.
  const int hg = lane >> 4;
  const bool col0 = (m == 0);
  const float* xr = x + (size_t)(row0 + m) * (size_t)C;
  v8f c = {};
  for (int k0 = 0; k0 < C; k0 += 32) {
    v16h a, bm;
#pragma unroll
    for (int e = 0; e < 16; ++e) {
      int v  = e >> 1;
      int Ka = ((v >> 2) << 4) + (hg << 3) + ((v & 3) << 1) + (e & 1);
      a[e]  = (_Float16)xr[k0 + Ka];
      int Kb = (hg << 4) + e;                 // B 32x16: lanes0-15 K=0..15
      float wv = W[k0 + Kb];
      bm[e] = col0 ? (_Float16)wv : (_Float16)0.0f;
    }
    c = __builtin_amdgcn_wmma_f32_16x16x32_f16(false, a, false, bm,
                                               (short)0, c, false, false);
  }
  if (lane == 0) {
#pragma unroll
    for (int r = 0; r < 8; ++r) xw[row0 + r] = c[r];
  } else if (lane == 16) {
#pragma unroll
    for (int r = 0; r < 8; ++r) xw[row0 + 8 + r] = c[r];
  }
#else
  // Host pass / no WMMA: plain dot product (never the device path on gfx1250).
  if (lane < 16) {
    const float* xr = x + (size_t)(row0 + m) * (size_t)C;
    float s = 0.0f;
    for (int k = 0; k < C; ++k) s += xr[k] * W[k];
    xw[row0 + m] = s;
  }
#endif
}

// ------------------------------------------------------------ edge work ----
__global__ void deg_kernel(const int* __restrict__ dst,
                           unsigned* __restrict__ cnt, int E) {
  int e = blockIdx.x * blockDim.x + threadIdx.x;
  if (e < E) atomicAdd(&cnt[dst[e]], 1u);
}

__global__ void dis_kernel(const unsigned* __restrict__ cnt,
                           float* __restrict__ dis, int N) {
  int i = blockIdx.x * blockDim.x + threadIdx.x;
  if (i < N) dis[i] = rsqrtf(1.0f + (float)cnt[i]);   // self loop included
}

// Deterministic segment-sum: Q32.32 fixed-point accumulate via u64 atomics.
__global__ void agg_kernel(const int* __restrict__ src, const int* __restrict__ dst,
                           const float* __restrict__ dis, const float* __restrict__ xw,
                           unsigned long long* __restrict__ aggI, int E) {
  int e = blockIdx.x * blockDim.x + threadIdx.x;
  if (e < E) {
    int s = src[e], d = dst[e];
    float v = dis[s] * dis[d] * xw[s];
    long long q = (long long)llrintf(v * 4294967296.0f);
    atomicAdd(&aggI[d], (unsigned long long)q);
  }
}

__global__ void score_kernel(const unsigned long long* __restrict__ aggI,
                             const float* __restrict__ dis,
                             const float* __restrict__ xw,
                             const float* __restrict__ b,
                             float* __restrict__ score,
                             unsigned* __restrict__ keys,
                             unsigned* __restrict__ idx, int N) {
  int i = blockIdx.x * blockDim.x + threadIdx.x;
  if (i < N) {
    float a  = (float)((double)(long long)aggI[i] * (1.0 / 4294967296.0));
    float sc = a + dis[i] * dis[i] * xw[i] + b[0];
    score[i] = sc;
    unsigned u = __float_as_uint(sc);
    unsigned o = (u & 0x80000000u) ? ~u : (u | 0x80000000u); // order-preserving
    keys[i] = ~o;            // ascending sort on key == descending on score
    idx[i]  = (unsigned)i;
  }
}

// ---------------------------------------------- stable LSD radix sort ------
__global__ void radix_hist(const unsigned* __restrict__ keys,
                           unsigned* __restrict__ blockHist,
                           int n, int nTiles, int shift) {
  __shared__ unsigned h[256];
  h[threadIdx.x] = 0u;
  __syncthreads();
  int base = blockIdx.x * RS_TILE;
  for (int s = 0; s < RS_ITEMS; ++s) {
    int i = base + s * RS_BLOCK + threadIdx.x;
    if (i < n) atomicAdd(&h[(keys[i] >> shift) & 255u], 1u);
  }
  __syncthreads();
  blockHist[(size_t)threadIdx.x * nTiles + blockIdx.x] = h[threadIdx.x];
}

__global__ void radix_scan(unsigned* __restrict__ blockHist,
                           unsigned* __restrict__ digitBase, int nTiles) {
  __shared__ unsigned tot[256];
  int d = threadIdx.x;
  unsigned run = 0;
  for (int t = 0; t < nTiles; ++t) {
    unsigned v = blockHist[(size_t)d * nTiles + t];
    blockHist[(size_t)d * nTiles + t] = run;   // exclusive within digit
    run += v;
  }
  tot[d] = run;
  __syncthreads();
  unsigned base = 0;
  for (int j = 0; j < d; ++j) base += tot[j];  // exclusive scan over digits
  digitBase[d] = base;
}

__global__ void radix_scatter(const unsigned* __restrict__ keysIn,
                              const unsigned* __restrict__ idxIn,
                              unsigned* __restrict__ keysOut,
                              unsigned* __restrict__ idxOut,
                              const unsigned* __restrict__ blockHist,
                              const unsigned* __restrict__ digitBase,
                              int n, int nTiles, int shift) {
  __shared__ unsigned cnt[256];    // per-digit running count within tile
  __shared__ unsigned dl[RS_BLOCK];
  int tid = threadIdx.x;
  cnt[tid] = 0u;
  __syncthreads();
  int base = blockIdx.x * RS_TILE;
  for (int s = 0; s < RS_ITEMS; ++s) {
    int i = base + s * RS_BLOCK + tid;
    unsigned key = 0, id = 0, d = 0x100u;
    bool valid = (i < n);
    if (valid) { key = keysIn[i]; id = idxIn[i]; d = (key >> shift) & 255u; }
    dl[tid] = d;
    __syncthreads();
    if (valid) {
      unsigned r = cnt[d];                       // earlier sub-batches
      for (int j = 0; j < tid; ++j) r += (dl[j] == d);  // stable in-batch rank
      unsigned pos = digitBase[d] + blockHist[(size_t)d * nTiles + blockIdx.x] + r;
      keysOut[pos] = key;
      idxOut[pos]  = id;
    }
    __syncthreads();
    if (valid) atomicAdd(&cnt[d], 1u);
    __syncthreads();
  }
}

// ----------------------------------------------------------- selection ----
__global__ void select_kernel(const unsigned* __restrict__ idxSorted,
                              const float* __restrict__ score,
                              int* __restrict__ perm_out,
                              int* __restrict__ new_idx,
                              float* __restrict__ gate, int K) {
  int i = blockIdx.x * blockDim.x + threadIdx.x;
  if (i < K) {
    int node = (int)idxSorted[i];
    perm_out[i]   = node;
    new_idx[node] = i;
    gate[i] = tanhf(score[node]);
  }
}

// x_out[i,:] = x[perm[i],:] * gate[i]  (block per row, float4 streams)
__global__ void gather_kernel(const float* __restrict__ x,
                              const int* __restrict__ perm,
                              const float* __restrict__ gate,
                              float* __restrict__ xout, int C4) {
  int i    = blockIdx.x;
  int node = perm[i];
  float g  = gate[i];
  const float4* s = (const float4*)(x    + (size_t)node * (size_t)(C4 * 4));
  float4*       o = (float4*)      (xout + (size_t)i    * (size_t)(C4 * 4));
  for (int t = threadIdx.x; t < C4; t += blockDim.x) {
    float4 v = s[t];
    v.x *= g; v.y *= g; v.z *= g; v.w *= g;
    o[t] = v;
  }
}

__global__ void filter_kernel(const int* __restrict__ src, const int* __restrict__ dst,
                              const int* __restrict__ new_idx,
                              int* __restrict__ out_src, int* __restrict__ out_dst,
                              unsigned char* __restrict__ valid, int E) {
  int e = blockIdx.x * blockDim.x + threadIdx.x;
  if (e < E) {
    int ns = new_idx[src[e]];
    int nd = new_idx[dst[e]];
    bool v = (ns >= 0) && (nd >= 0);
    out_src[e] = v ? ns : -1;
    out_dst[e] = v ? nd : -1;
    valid[e]   = (unsigned char)(v ? 1 : 0);
  }
}

// -------------------------------------------------------------- launch ----
extern "C" void kernel_launch(void* const* d_in, const int* in_sizes, int n_in,
                              void* d_out, int out_size, void* d_ws, size_t ws_size,
                              hipStream_t stream) {
  const float* x          = (const float*)d_in[0];
  const int*   edge_index = (const int*)d_in[1];
  const float* W          = (const float*)d_in[2];
  const float* b          = (const float*)d_in[3];
  const int C = in_sizes[2];          // 256
  const int N = in_sizes[0] / C;      // 500000
  const int E = in_sizes[1] / 2;      // 16000000
  const int K = (N + 1) / 2;          // ceil(0.5*N)
  const int* esrc = edge_index;
  const int* edst = edge_index + (size_t)E;

  // ---- workspace carve (256B aligned blocks, ~23 MB total) ----
  uintptr_t w = (uintptr_t)d_ws;
  auto carve = [&](size_t bytes) -> void* {
    void* p = (void*)w;
    w += (bytes + 255) & ~(size_t)255;
    return p;
  };
  float*    xw    = (float*)carve((size_t)N * 4);
  float*    dis   = (float*)carve((size_t)N * 4);
  float*    score = (float*)carve((size_t)N * 4);
  unsigned* cnt   = (unsigned*)carve((size_t)N * 4);
  unsigned long long* aggI = (unsigned long long*)carve((size_t)N * 8);
  unsigned* keysA = (unsigned*)carve((size_t)N * 4);
  unsigned* keysB = (unsigned*)carve((size_t)N * 4);
  unsigned* idxA  = (unsigned*)carve((size_t)N * 4);
  unsigned* idxB  = (unsigned*)carve((size_t)N * 4);
  int*      new_idx = (int*)carve((size_t)N * 4);
  float*    gate  = (float*)carve((size_t)K * 4);
  const int nTiles = (N + RS_TILE - 1) / RS_TILE;
  unsigned* blockHist = (unsigned*)carve((size_t)nTiles * 256 * 4);
  unsigned* digitBase = (unsigned*)carve(256 * 4);

  // ---- output carve (flat concat in reference return order) ----
  float*         x_out   = (float*)d_out;
  int*           new_src = (int*)((char*)d_out + (size_t)K * (size_t)C * 4);
  int*           new_dst = new_src + (size_t)E;
  unsigned char* evalid  = (unsigned char*)(new_dst + (size_t)E);
  int*           permO   = (int*)(evalid + (size_t)E);

  int nbN = (N + 255) / 256;
  int nbE = (E + 255) / 256;
  int nbK = (K + 255) / 256;

  init_kernel<<<nbN, 256, 0, stream>>>(cnt, aggI, new_idx, N);

  {
    int waves  = (N + 15) / 16;
    int blocks = (waves + 7) / 8;    // 256 threads = 8 waves/block
    xw_score_kernel<<<blocks, 256, 0, stream>>>(x, W, xw, N, C);
  }

  deg_kernel<<<nbE, 256, 0, stream>>>(edst, cnt, E);
  dis_kernel<<<nbN, 256, 0, stream>>>(cnt, dis, N);
  agg_kernel<<<nbE, 256, 0, stream>>>(esrc, edst, dis, xw, aggI, E);
  score_kernel<<<nbN, 256, 0, stream>>>(aggI, dis, xw, b, score, keysA, idxA, N);

  // 4-pass stable LSD radix sort, ascending on ~orderable(score) => descending score
  unsigned *ki = keysA, *ii = idxA, *ko = keysB, *io = idxB;
  for (int p = 0; p < 4; ++p) {
    int shift = p * 8;
    radix_hist   <<<nTiles, 256, 0, stream>>>(ki, blockHist, N, nTiles, shift);
    radix_scan   <<<1,      256, 0, stream>>>(blockHist, digitBase, nTiles);
    radix_scatter<<<nTiles, 256, 0, stream>>>(ki, ii, ko, io, blockHist, digitBase,
                                              N, nTiles, shift);
    unsigned* t;
    t = ki; ki = ko; ko = t;
    t = ii; ii = io; io = t;
  }
  // after 4 swaps, sorted result is back in keysA/idxA (== ki/ii)

  select_kernel<<<nbK, 256, 0, stream>>>(ii, score, permO, new_idx, gate, K);
  gather_kernel<<<K, 64, 0, stream>>>(x, permO, gate, x_out, C / 4);
  filter_kernel<<<nbE, 256, 0, stream>>>(esrc, edst, new_idx, new_src, new_dst,
                                         evalid, E);
}
